// KlcpdGenTl_83451214561283
// MI455X (gfx1250) — matrix operand order repaired and checked by reading, not compiled, not run on hardware
//
#include <hip/hip_runtime.h>

// ---------------------------------------------------------------------------
// KL-CPD generator (embed -> GRU encoder -> +noise -> GRU decoder -> linear)
// MI455X / gfx1250, wave32.
//
// - All GEMMs on v_wmma_f32_16x16x32_f16 (f32 accumulate).
// - Weights pre-transposed to [N][K] f16 so every LDS tile stage is a
//   contiguous 16B copy -> GLOBAL_LOAD_ASYNC_TO_LDS_B128 (ASYNCcnt), with
//   double-buffered LDS and s_wait_asynccnt for copy/compute overlap.
// - GRU step fuses x@W_ih and h@W_hh into shared accumulators (r,z fused;
//   xn,hn separate for the PyTorch n-gate) plus the full cell epilogue.
//   Phase loops are split so accumulator routing is compile-time (no
//   cndmask select chains around the WMMAs).
// ---------------------------------------------------------------------------

typedef __attribute__((ext_vector_type(16))) _Float16 v16h;
typedef __attribute__((ext_vector_type(8)))  _Float16 v8h;
typedef __attribute__((ext_vector_type(8)))  float    v8f;
typedef __attribute__((ext_vector_type(4)))  int      v4i;

constexpr int T  = 64;
constexpr int B  = 1024;
constexpr int D  = 128;
constexpr int E  = 256;
constexpr int H  = 512;
constexpr int TB = T * B;
constexpr int H3 = 3 * H;

// ----------------------- async global->LDS (CDNA5) -------------------------
#if defined(__gfx1250__) && __has_builtin(__builtin_amdgcn_global_load_async_to_lds_b128)
#define USE_ASYNC_LDS 1
typedef __attribute__((address_space(1))) v4i gv4i;   // global int4
typedef __attribute__((address_space(3))) v4i lv4i;   // LDS int4
#else
#define USE_ASYNC_LDS 0
#endif

__device__ __forceinline__ void async_copy16(const _Float16* g, _Float16* l) {
#if USE_ASYNC_LDS
  __builtin_amdgcn_global_load_async_to_lds_b128((gv4i*)g, (lv4i*)l, 0, 0);
#else
  *(v8h*)l = *(const v8h*)g;
#endif
}

__device__ __forceinline__ void async_wait() {
#if USE_ASYNC_LDS && __has_builtin(__builtin_amdgcn_s_wait_asynccnt)
  __builtin_amdgcn_s_wait_asynccnt(0);
#endif
}

// ------------------------- WMMA fragment helpers ---------------------------
// ISA 7.12.2 layouts (wave32):
//  A (16x32 f16): lane m=L&15, hi=L>>4; halves 0..7 -> k=8*hi+0..7,
//                 halves 8..15 -> k=16+8*hi+0..7.
//  B (32x16 f16): lane n=L&15, hi=L>>4; same k order as A  (LDS holds B^T,
//                 so each lane reads two 8-half runs).
//  C/D (16x16 f32): lane n=L&15; VGPR v -> m=v+8*(L>>4).

__device__ __forceinline__ v16h frag_a(const _Float16* sA, int row0, int ld, int lane) {
  int m = lane & 15, hi = lane >> 4;
  const _Float16* p = sA + (size_t)(row0 + m) * ld + 8 * hi;
  v8h lo = *(const v8h*)p;
  v8h up = *(const v8h*)(p + 16);
  v16h a;
#pragma unroll
  for (int i = 0; i < 8; ++i) { a[i] = lo[i]; a[i + 8] = up[i]; }
  return a;
}

__device__ __forceinline__ v16h frag_bT(const _Float16* sBT, int col0, int ld, int lane) {
  int n = lane & 15, hi = lane >> 4;
  const _Float16* p = sBT + (size_t)(col0 + n) * ld + 8 * hi;
  v8h x = *(const v8h*)p;
  v8h y = *(const v8h*)(p + 16);
  v16h b;
#pragma unroll
  for (int i = 0; i < 8; ++i) { b[i] = x[i]; b[i + 8] = y[i]; }
  return b;
}

__device__ __forceinline__ v8f wmma_f16(v16h a, v16h b, v8f c) {
  return __builtin_amdgcn_wmma_f32_16x16x32_f16(false, a, false, b, (short)0, c,
                                                false, false);
}

__device__ __forceinline__ float sigmoidf_fast(float x) {
  return 1.0f / (1.0f + __expf(-x));
}
__device__ __forceinline__ float tanhf_fast(float x) {
  return 2.0f / (1.0f + __expf(-2.0f * x)) - 1.0f;
}

// One fused GRU MMA iteration: gates r,z -> acc[0],acc[1]; n-gate -> acc[AN]
// (AN=2 during the x phase, AN=3 during the h phase; compile-time).
template <int AN>
__device__ __forceinline__ void gru_mma(const _Float16* sAc, const _Float16* sBTc,
                                        int wm, int wn, int lane, v8f (&acc)[4][2]) {
  v16h af = frag_a(sAc, wm * 16, 40, lane);
#pragma unroll
  for (int g = 0; g < 3; ++g) {
    const int ai = (g < 2) ? g : AN;   // compile-time after unroll
#pragma unroll
    for (int nt = 0; nt < 2; ++nt) {
      v16h bf = frag_bT(sBTc + g * (64 * 40), wn * 32 + nt * 16, 40, lane);
      acc[ai][nt] = wmma_f16(af, bf, acc[ai][nt]);
    }
  }
}

// --------------------------- generic WMMA GEMM -----------------------------
// C[M,N] = act(A[M,K]_f16 @ B[K,N]_f16 + bias[N]), B given TRANSPOSED
// (BT[N][K]). M%128==0, N%128==0, K%32==0. 256 threads = 8 waves (2x4);
// wave computes 64x32 (4x2 tiles). Double-buffered async LDS staging.
template <bool RELU, bool OUT_F16>
__global__ __launch_bounds__(256)
void wmma_gemm(const _Float16* __restrict__ A, const _Float16* __restrict__ BT,
               const float* __restrict__ bias, void* __restrict__ Cout,
               int M, int N, int K) {
  __shared__ __align__(16) _Float16 sA[2][128][40];
  __shared__ __align__(16) _Float16 sBT[2][128][40];

  const int tid  = threadIdx.x;
  const int lane = tid & 31;
  const int w    = tid >> 5;
  const int wm   = w >> 2;          // 0..1 -> 64 rows
  const int wn   = w & 3;           // 0..3 -> 32 cols
  const int r0   = blockIdx.x * 128;
  const int c0   = blockIdx.y * 128;

  const int arow = tid >> 1;              // 0..127
  const int acol = (tid & 1) * 16;        // 0 / 16
  const int bcol = tid & 127;             // 0..127
  const int bkh  = (tid >> 7) * 16;       // 0 / 16

  v8f acc[4][2];
#pragma unroll
  for (int mt = 0; mt < 4; ++mt)
#pragma unroll
    for (int nt = 0; nt < 2; ++nt) acc[mt][nt] = (v8f){};

  auto stage = [&](int buf, int k0) {
    const _Float16* ga = A + (size_t)(r0 + arow) * K + k0 + acol;
    async_copy16(ga,     &sA[buf][arow][acol]);
    async_copy16(ga + 8, &sA[buf][arow][acol + 8]);
    const _Float16* gb = BT + (size_t)(c0 + bcol) * K + k0 + bkh;
    async_copy16(gb,     &sBT[buf][bcol][bkh]);
    async_copy16(gb + 8, &sBT[buf][bcol][bkh + 8]);
  };

  stage(0, 0);
  async_wait();
  __syncthreads();

  const int iters = K >> 5;
  for (int it = 0; it < iters; ++it) {
    const int cur = it & 1;
    if (it + 1 < iters) stage(cur ^ 1, (it + 1) * 32);  // overlap with WMMAs

    v16h af[4];
#pragma unroll
    for (int mt = 0; mt < 4; ++mt)
      af[mt] = frag_a(&sA[cur][0][0], wm * 64 + mt * 16, 40, lane);
    v16h bf[2];
#pragma unroll
    for (int nt = 0; nt < 2; ++nt)
      bf[nt] = frag_bT(&sBT[cur][0][0], wn * 32 + nt * 16, 40, lane);
#pragma unroll
    for (int mt = 0; mt < 4; ++mt)
#pragma unroll
      for (int nt = 0; nt < 2; ++nt)
        acc[mt][nt] = wmma_f16(af[mt], bf[nt], acc[mt][nt]);

    async_wait();
    __syncthreads();
  }

  const int hi = lane >> 4, ln = lane & 15;
#pragma unroll
  for (int nt = 0; nt < 2; ++nt) {
    int n = c0 + wn * 32 + nt * 16 + ln;
    float bv = bias[n];
#pragma unroll
    for (int mt = 0; mt < 4; ++mt) {
#pragma unroll
      for (int v = 0; v < 8; ++v) {
        int m = r0 + wm * 64 + mt * 16 + v + 8 * hi;
        float x = acc[mt][nt][v] + bv;
        if (RELU) x = fmaxf(x, 0.0f);
        size_t idx = (size_t)m * N + n;
        if constexpr (OUT_F16) ((_Float16*)Cout)[idx] = (_Float16)x;
        else                   ((float*)Cout)[idx]    = x;
      }
    }
  }
}

// ------------------------ fused GRU step kernel ----------------------------
// r = sig(xr+hr+b), z = sig(xz+hz+b), n = tanh(xn+bihn + r*(hn+bhhn)),
// h' = (1-z)*n + z*h.  Grid (B/64, H/64); 256 threads = 8 waves (4x2).
// Accumulators: 0=r (x+h fused), 1=z (fused), 2=xn (x phase), 3=hn (h phase).
// Weights are pre-transposed: WihT[3H][E], WhhT[3H][H].
__global__ __launch_bounds__(256)
void gru_step(const _Float16* __restrict__ x_t,   // [B,E] f16 or nullptr (x==0)
              const _Float16* __restrict__ h16,   // [B,H] f16 (prev)
              const float*   __restrict__ h32,    // [B,H] f32 (prev)
              const _Float16* __restrict__ WihT,  // [3H,E] f16
              const _Float16* __restrict__ WhhT,  // [3H,H] f16
              const float* __restrict__ bih, const float* __restrict__ bhh,
              float* __restrict__ h32_out, _Float16* __restrict__ h16_out,
              _Float16* __restrict__ ys_out)      // [B,H] f16 or nullptr
{
  __shared__ __align__(16) _Float16 sA[2][64][40];
  __shared__ __align__(16) _Float16 sBT[2][3][64][40];

  const int tid  = threadIdx.x;
  const int lane = tid & 31;
  const int w    = tid >> 5;
  const int wm   = w >> 1;          // 0..3 -> 16 rows
  const int wn   = w & 1;           // 0..1 -> 32 cols
  const int m0   = blockIdx.x * 64;
  const int j0   = blockIdx.y * 64;

  const int arow = tid >> 2;            // 0..63
  const int acol = (tid & 3) * 8;       // 0..24
  const int bc   = tid & 63;            // 0..63
  const int bkh  = (tid >> 6) * 8;      // 0,8,16,24

  v8f acc[4][2];
#pragma unroll
  for (int g = 0; g < 4; ++g)
#pragma unroll
    for (int nt = 0; nt < 2; ++nt) acc[g][nt] = (v8f){};

  const int xIters = x_t ? (E >> 5) : 0;   // 8 or 0
  const int hIters = H >> 5;               // 16

  // explicit pointers per phase -> no runtime pointer selects
  auto stage = [&](int buf, const _Float16* Ag, const _Float16* Wg, int Kp, int k0) {
    async_copy16(Ag + (size_t)(m0 + arow) * Kp + k0 + acol, &sA[buf][arow][acol]);
#pragma unroll
    for (int g = 0; g < 3; ++g)
      async_copy16(Wg + (size_t)(g * H + j0 + bc) * Kp + k0 + bkh,
                   &sBT[buf][g][bc][bkh]);
  };

  if (xIters > 0) stage(0, x_t, WihT, E, 0);
  else            stage(0, h16, WhhT, H, 0);
  async_wait();
  __syncthreads();

  int cur = 0;
  // ---- x phase: n-gate -> acc[2] ----
  for (int i = 0; i < xIters; ++i) {
    if (i + 1 < xIters) stage(cur ^ 1, x_t, WihT, E, (i + 1) * 32);
    else                stage(cur ^ 1, h16, WhhT, H, 0);  // cross-phase prefetch
    gru_mma<2>(&sA[cur][0][0], &sBT[cur][0][0][0], wm, wn, lane, acc);
    async_wait();
    __syncthreads();
    cur ^= 1;
  }
  // ---- h phase: n-gate -> acc[3] ----
  for (int i = 0; i < hIters; ++i) {
    if (i + 1 < hIters) stage(cur ^ 1, h16, WhhT, H, (i + 1) * 32);
    gru_mma<3>(&sA[cur][0][0], &sBT[cur][0][0][0], wm, wn, lane, acc);
    async_wait();
    __syncthreads();
    cur ^= 1;
  }

  // -------- GRU cell epilogue --------
  const int hi = lane >> 4, ln = lane & 15;
#pragma unroll
  for (int nt = 0; nt < 2; ++nt) {
    int j = j0 + wn * 32 + nt * 16 + ln;
    float br  = bih[0 * H + j] + bhh[0 * H + j];
    float bz  = bih[1 * H + j] + bhh[1 * H + j];
    float bxn = bih[2 * H + j];
    float bhn = bhh[2 * H + j];
#pragma unroll
    for (int v = 0; v < 8; ++v) {
      int m = m0 + wm * 16 + v + 8 * hi;
      size_t idx = (size_t)m * H + j;
      float r = sigmoidf_fast(acc[0][nt][v] + br);
      float z = sigmoidf_fast(acc[1][nt][v] + bz);
      float n = tanhf_fast(acc[2][nt][v] + bxn + r * (acc[3][nt][v] + bhn));
      float hn = (1.0f - z) * n + z * h32[idx];
      h32_out[idx] = hn;
      _Float16 hh = (_Float16)hn;
      h16_out[idx] = hh;
      if (ys_out) ys_out[idx] = hh;
    }
  }
}

// --------------------------- elementwise helpers ---------------------------
__global__ __launch_bounds__(256)
void cvt_f32_f16(const float* __restrict__ s, _Float16* __restrict__ d, int n) {
  int i = blockIdx.x * 256 + threadIdx.x;
  if (i < n) d[i] = (_Float16)s[i];
}

// out f16 [N][K] = transpose(in f32 [K][N]); K,N multiples of 32.
__global__ __launch_bounds__(256)
void transpose_cvt(const float* __restrict__ in, _Float16* __restrict__ outp,
                   int K, int N) {
  __shared__ float tile[32][33];
  int k0 = blockIdx.x * 32, n0 = blockIdx.y * 32;
  int tx = threadIdx.x & 31, ty = threadIdx.x >> 5;   // ty 0..7
#pragma unroll
  for (int i = ty; i < 32; i += 8)
    tile[i][tx] = in[(size_t)(k0 + i) * N + n0 + tx];
  __syncthreads();
#pragma unroll
  for (int i = ty; i < 32; i += 8)
    outp[(size_t)(n0 + i) * K + k0 + tx] = (_Float16)tile[tx][i];
}

__global__ __launch_bounds__(256)
void zero_h(float* __restrict__ h32, _Float16* __restrict__ h16, int n) {
  int i = blockIdx.x * 256 + threadIdx.x;
  if (i < n) { h32[i] = 0.0f; h16[i] = (_Float16)0.0f; }
}

__global__ __launch_bounds__(256)
void add_noise(const float* __restrict__ h, const float* __restrict__ nz,
               float* __restrict__ o32, _Float16* __restrict__ o16, int n) {
  int i = blockIdx.x * 256 + threadIdx.x;
  if (i < n) {
    float v = h[i] + nz[i];
    o32[i] = v;
    o16[i] = (_Float16)v;
  }
}

// ------------------------------- launcher ----------------------------------
extern "C" void kernel_launch(void* const* d_in, const int* in_sizes, int n_in,
                              void* d_out, int out_size, void* d_ws, size_t ws_size,
                              hipStream_t stream) {
  (void)in_sizes; (void)n_in; (void)out_size; (void)ws_size;
  const float* past   = (const float*)d_in[0];
  const float* fut    = (const float*)d_in[1];
  const float* noise  = (const float*)d_in[2];
  const float* W_emb  = (const float*)d_in[3];
  const float* b_emb  = (const float*)d_in[4];
  const float* W_ih_e = (const float*)d_in[5];
  const float* W_hh_e = (const float*)d_in[6];
  const float* b_ih_e = (const float*)d_in[7];
  const float* b_hh_e = (const float*)d_in[8];
  const float* W_ih_d = (const float*)d_in[9];
  const float* W_hh_d = (const float*)d_in[10];
  const float* b_ih_d = (const float*)d_in[11];
  const float* b_hh_d = (const float*)d_in[12];
  const float* W_out  = (const float*)d_in[13];
  const float* b_out  = (const float*)d_in[14];
  float* out = (float*)d_out;

  // workspace carve (256B aligned)
  char* p = (char*)d_ws;
  auto take = [&](size_t bytes) -> char* {
    char* r = p;
    p += (bytes + 255) & ~(size_t)255;
    return r;
  };
  _Float16* past16   = (_Float16*)take((size_t)TB * D * 2);
  _Float16* fut16    = (_Float16*)take((size_t)TB * D * 2);
  _Float16* WembT16  = (_Float16*)take((size_t)E * D * 2);    // [E][D]
  _Float16* WihET16  = (_Float16*)take((size_t)H3 * E * 2);   // [3H][E]
  _Float16* WihDT16  = (_Float16*)take((size_t)H3 * E * 2);
  _Float16* WhhET16  = (_Float16*)take((size_t)H3 * H * 2);   // [3H][H]
  _Float16* WhhDT16  = (_Float16*)take((size_t)H3 * H * 2);
  _Float16* WoutT16  = (_Float16*)take((size_t)D * H * 2);    // [D][H]
  _Float16* embE16   = (_Float16*)take((size_t)TB * E * 2);
  _Float16* embD16   = (_Float16*)take((size_t)TB * E * 2);
  _Float16* ys16     = (_Float16*)take((size_t)TB * H * 2);
  float*    hA32     = (float*)take((size_t)B * H * 4);
  float*    hB32     = (float*)take((size_t)B * H * 4);
  _Float16* hA16     = (_Float16*)take((size_t)B * H * 2);
  _Float16* hB16     = (_Float16*)take((size_t)B * H * 2);

  // activation f32 -> f16
  cvt_f32_f16<<<(TB * D + 255) / 256, 256, 0, stream>>>(past, past16, TB * D);
  cvt_f32_f16<<<(TB * D + 255) / 256, 256, 0, stream>>>(fut,  fut16,  TB * D);

  // weight f32 [K][N] -> f16 transposed [N][K]
  auto tc = [&](const float* s, _Float16* d, int K, int N) {
    transpose_cvt<<<dim3(K / 32, N / 32), 256, 0, stream>>>(s, d, K, N);
  };
  tc(W_emb,  WembT16, D, E);
  tc(W_ih_e, WihET16, E, H3);
  tc(W_ih_d, WihDT16, E, H3);
  tc(W_hh_e, WhhET16, H, H3);
  tc(W_hh_d, WhhDT16, H, H3);
  tc(W_out,  WoutT16, H, D);

  // embeddings: relu(x @ W_emb + b_emb) -> f16
  wmma_gemm<true, true><<<dim3(TB / 128, E / 128), 256, 0, stream>>>(
      past16, WembT16, b_emb, embE16, TB, E, D);
  wmma_gemm<true, true><<<dim3(TB / 128, E / 128), 256, 0, stream>>>(
      fut16, WembT16, b_emb, embD16, TB, E, D);

  // encoder: h0 = 0, 64 fused steps (ping-pong A<->B, final lands in A)
  zero_h<<<(B * H + 255) / 256, 256, 0, stream>>>(hA32, hA16, B * H);
  dim3 sgrid(B / 64, H / 64);
  for (int t = 0; t < T; ++t) {
    bool even = (t & 1) == 0;
    gru_step<<<sgrid, 256, 0, stream>>>(
        embE16 + (size_t)t * B * E,
        even ? hA16 : hB16, even ? hA32 : hB32,
        WihET16, WhhET16, b_ih_e, b_hh_e,
        even ? hB32 : hA32, even ? hB16 : hA16, nullptr);
  }

  // hidden = h_enc + noise  (into B buffers)
  add_noise<<<(B * H + 255) / 256, 256, 0, stream>>>(hA32, noise, hB32, hB16, B * H);

  // decoder: input shifted (t=0 -> zeros), emits ys
  for (int t = 0; t < T; ++t) {
    bool even = (t & 1) == 0;  // even steps read B (hidden starts in B)
    const _Float16* xt = (t == 0) ? nullptr : embD16 + (size_t)(t - 1) * B * E;
    gru_step<<<sgrid, 256, 0, stream>>>(
        xt,
        even ? hB16 : hA16, even ? hB32 : hA32,
        WihDT16, WhhDT16, b_ih_d, b_hh_d,
        even ? hA32 : hB32, even ? hA16 : hB16,
        ys16 + (size_t)t * B * H);
  }

  // output projection: out = ys @ W_out + b_out  (f32)
  wmma_gemm<false, false><<<dim3(TB / 128, D / 128), 256, 0, stream>>>(
      ys16, WoutT16, b_out, out, TB, D, H);
}
